// UnoptimizedNeuralNet_6064493822319
// MI455X (gfx1250) — matrix operand order repaired and checked by reading, not compiled, not run on hardware
//
#include <hip/hip_runtime.h>
#include <hip/hip_bf16.h>

// ---------------------------------------------------------------------------
// MLP: hidden = relu(x @ W1^T + b1); out = softmax(hidden @ W2^T + b2)
// x:[8192,4096] W1:[4096,4096] W2:[1000,4096]  all f32 in, f32 out.
// Compute-bound (~342 GFLOP vs ~16us of HBM traffic at 23.3 TB/s), so the
// matrix-pipe precision dominates: bf16 WMMA (K=32/instr) with f32 accumulate.
// Global->LDS uses CDNA5 async-to-LDS (ASYNCcnt) so tile fills bypass the
// VALU/VGPR path entirely and overlap with the WMMA stream.
// ---------------------------------------------------------------------------

typedef __attribute__((ext_vector_type(16))) __bf16 v16bf;
typedef __attribute__((ext_vector_type(8)))  float  v8f;
typedef __attribute__((ext_vector_type(4)))  int    v4i;

union FragAB { v16bf v; uint4 u[2]; };   // 32 bytes: 16 bf16

constexpr int BM  = 128;   // block tile M
constexpr int BN  = 128;   // block tile N
constexpr int BK  = 32;    // block tile K (one bf16 WMMA consumes all of it)
constexpr int LDA = 40;    // LDS row stride in bf16 (32+8 pad, 80B keeps 16B align)

#if defined(__has_builtin)
#  if __has_builtin(__builtin_amdgcn_global_load_async_to_lds_b128) && \
      __has_builtin(__builtin_amdgcn_s_wait_asynccnt)
#    define USE_ASYNC_LDS 1
#  endif
#endif
#ifndef USE_ASYNC_LDS
#  define USE_ASYNC_LDS 0
#endif

#if USE_ASYNC_LDS
typedef __attribute__((address_space(1))) v4i GV4;   // global v4i
typedef __attribute__((address_space(3))) v4i LV4;   // LDS v4i
#endif

// ---------------------------------------------------------------------------
// f32 -> bf16 conversion, 8 elements per thread
// ---------------------------------------------------------------------------
__global__ __launch_bounds__(256)
void cvt8_f32_bf16(const float* __restrict__ in, __bf16* __restrict__ out, size_t n8) {
    size_t i = (size_t)blockIdx.x * blockDim.x + threadIdx.x;
    if (i >= n8) return;
    float4 f0 = ((const float4*)in)[i * 2 + 0];
    float4 f1 = ((const float4*)in)[i * 2 + 1];
    union { __bf16 h[8]; uint4 u; } p;
    p.h[0] = (__bf16)f0.x; p.h[1] = (__bf16)f0.y;
    p.h[2] = (__bf16)f0.z; p.h[3] = (__bf16)f0.w;
    p.h[4] = (__bf16)f1.x; p.h[5] = (__bf16)f1.y;
    p.h[6] = (__bf16)f1.z; p.h[7] = (__bf16)f1.w;
    ((uint4*)out)[i] = p.u;
}

// Same, but pads rows [rowsIn, rowsOut) with zeros (W2: 1000 -> 1024 rows).
__global__ __launch_bounds__(256)
void cvt8_f32_bf16_padrows(const float* __restrict__ in, __bf16* __restrict__ out,
                           int k8, int rowsIn, int rowsOut) {
    size_t i = (size_t)blockIdx.x * blockDim.x + threadIdx.x;
    size_t total = (size_t)rowsOut * k8;
    if (i >= total) return;
    int row = (int)(i / k8);
    int c8  = (int)(i % k8);
    union { __bf16 h[8]; uint4 u; } p;
    if (row < rowsIn) {
        const float* src = in + (size_t)row * (k8 * 8) + c8 * 8;
        float4 f0 = ((const float4*)src)[0];
        float4 f1 = ((const float4*)src)[1];
        p.h[0] = (__bf16)f0.x; p.h[1] = (__bf16)f0.y;
        p.h[2] = (__bf16)f0.z; p.h[3] = (__bf16)f0.w;
        p.h[4] = (__bf16)f1.x; p.h[5] = (__bf16)f1.y;
        p.h[6] = (__bf16)f1.z; p.h[7] = (__bf16)f1.w;
    } else {
        p.u = make_uint4(0u, 0u, 0u, 0u);
    }
    ((uint4*)out)[i] = p.u;
}

// ---------------------------------------------------------------------------
// Tiled bf16 WMMA GEMM:  C[M,N] = A[M,K] * B[N,K]^T   (B rows are N, K-contig)
// 128 threads = 4 waves in a 2x2 grid; each wave owns a 64x64 patch =
// 4x4 tiles of 16x16 (16 v8f accumulators). 16 WMMAs per 16 ds_load_b128 per
// K-step. Double-buffered LDS filled by async-to-LDS copies.
// Epilogue: +bias, optional ReLU, store bf16 (hidden) or f32 (logits, n<nValid).
// ---------------------------------------------------------------------------
template<bool RELU, bool OUTBF16>
__global__ __launch_bounds__(128)
void wmma_gemm_bf16(const __bf16* __restrict__ A, const __bf16* __restrict__ Bm,
                    const float* __restrict__ bias,
                    float* __restrict__ Cf, __bf16* __restrict__ Cb,
                    int K, int ldc, int nValid) {
    __shared__ __align__(16) __bf16 As[2][BM * LDA];
    __shared__ __align__(16) __bf16 Bs[2][BN * LDA];

    const int tid  = threadIdx.x;
    const int lane = tid & 31;
    const int wave = tid >> 5;
    const int wr   = wave & 1;          // -> M offset wr*64
    const int wc   = wave >> 1;         // -> N offset wc*64
    const int m0   = blockIdx.y * BM;
    const int n0   = blockIdx.x * BN;

    v8f acc[4][4];
    #pragma unroll
    for (int mt = 0; mt < 4; ++mt)
        #pragma unroll
        for (int nt = 0; nt < 4; ++nt)
            #pragma unroll
            for (int r = 0; r < 8; ++r) acc[mt][nt][r] = 0.0f;

    const int nK = K / BK;

#if USE_ASYNC_LDS
    // Tile copy: 512 chunks of 8 bf16 per 128x32 tile; 128 threads -> 4 each.
    auto copy_tile = [&](int buf, int kblk) {
        const size_t kb = (size_t)kblk * BK;
        #pragma unroll
        for (int i = 0; i < 4; ++i) {
            const int c = tid + (i << 7);
            const int m = c >> 2, kc = (c & 3) << 3;
            __builtin_amdgcn_global_load_async_to_lds_b128(
                (GV4*)(A + (size_t)(m0 + m) * K + kb + kc),
                (LV4*)&As[buf][m * LDA + kc], 0, 0);
            __builtin_amdgcn_global_load_async_to_lds_b128(
                (GV4*)(Bm + (size_t)(n0 + m) * K + kb + kc),
                (LV4*)&Bs[buf][m * LDA + kc], 0, 0);
        }
    };
    copy_tile(0, 0);
    __builtin_amdgcn_s_wait_asynccnt(0);
    __syncthreads();
#else
    const int cm0 = tid >> 2,          ck0 = (tid & 3) << 3;
    const int cm1 = (tid + 128) >> 2,  ck1 = ((tid + 128) & 3) << 3;
    const int cm2 = (tid + 256) >> 2,  ck2 = ((tid + 256) & 3) << 3;
    const int cm3 = (tid + 384) >> 2,  ck3 = ((tid + 384) & 3) << 3;
    uint4 ra[4], rb[4];
    auto gload = [&](int kblk) {
        const size_t kb = (size_t)kblk * BK;
        ra[0] = *(const uint4*)(A  + (size_t)(m0 + cm0) * K + kb + ck0);
        rb[0] = *(const uint4*)(Bm + (size_t)(n0 + cm0) * K + kb + ck0);
        ra[1] = *(const uint4*)(A  + (size_t)(m0 + cm1) * K + kb + ck1);
        rb[1] = *(const uint4*)(Bm + (size_t)(n0 + cm1) * K + kb + ck1);
        ra[2] = *(const uint4*)(A  + (size_t)(m0 + cm2) * K + kb + ck2);
        rb[2] = *(const uint4*)(Bm + (size_t)(n0 + cm2) * K + kb + ck2);
        ra[3] = *(const uint4*)(A  + (size_t)(m0 + cm3) * K + kb + ck3);
        rb[3] = *(const uint4*)(Bm + (size_t)(n0 + cm3) * K + kb + ck3);
    };
    auto lstore = [&](int buf) {
        *(uint4*)&As[buf][cm0 * LDA + ck0] = ra[0];
        *(uint4*)&Bs[buf][cm0 * LDA + ck0] = rb[0];
        *(uint4*)&As[buf][cm1 * LDA + ck1] = ra[1];
        *(uint4*)&Bs[buf][cm1 * LDA + ck1] = rb[1];
        *(uint4*)&As[buf][cm2 * LDA + ck2] = ra[2];
        *(uint4*)&Bs[buf][cm2 * LDA + ck2] = rb[2];
        *(uint4*)&As[buf][cm3 * LDA + ck3] = ra[3];
        *(uint4*)&Bs[buf][cm3 * LDA + ck3] = rb[3];
    };
    gload(0);
    lstore(0);
    __syncthreads();
#endif

    // A fragment (ISA 7.12.2, 16-bit A 16x32): lane<16 -> K {0..7,16..23},
    // lane>=16 -> K {8..15,24..31}, row M = lane%16.
    const int arow = lane & 15;
    const int akc  = (lane >> 4) << 3;   // 0 or 8
    // B fragment: lane<16 -> K {0..7,16..23}, lane>=16 -> K {8..15,24..31}, col N = lane%16.
    const int bcol = lane & 15;
    const int bkb  = (lane >> 4) << 3;   // 0 or 8

    for (int kb = 0; kb < nK; ++kb) {
        const int cur   = kb & 1;
        const bool more = (kb + 1 < nK);

#if USE_ASYNC_LDS
        if (more) copy_tile(cur ^ 1, kb + 1);   // async engine fills next buffer
#else
        if (more) gload(kb + 1);
#endif

        FragAB af[4], bfr[4];
        #pragma unroll
        for (int t = 0; t < 4; ++t) {
            const __bf16* sa = &As[cur][(wr * 64 + t * 16 + arow) * LDA + akc];
            af[t].u[0] = *(const uint4*)(sa);
            af[t].u[1] = *(const uint4*)(sa + 16);
            const __bf16* sb = &Bs[cur][(wc * 64 + t * 16 + bcol) * LDA + bkb];
            bfr[t].u[0] = *(const uint4*)(sb);
            bfr[t].u[1] = *(const uint4*)(sb + 16);
        }
        #pragma unroll
        for (int mt = 0; mt < 4; ++mt)
            #pragma unroll
            for (int nt = 0; nt < 4; ++nt)
                acc[mt][nt] = __builtin_amdgcn_wmma_f32_16x16x32_bf16(
                    false, af[mt].v, false, bfr[nt].v,
                    (short)0, acc[mt][nt], false, false);

#if USE_ASYNC_LDS
        if (more) __builtin_amdgcn_s_wait_asynccnt(0);
#else
        if (more) lstore(cur ^ 1);
#endif
        __syncthreads();
    }

    // epilogue: C/D layout — VGPR r, lane L: M = r + 8*(L>=16), N = L%16
    const int col  = lane & 15;
    const int rofs = (lane >> 4) << 3;
    #pragma unroll
    for (int mt = 0; mt < 4; ++mt) {
        #pragma unroll
        for (int nt = 0; nt < 4; ++nt) {
            const int n = n0 + wc * 64 + nt * 16 + col;
            const bool nok = (n < nValid);
            const float bv = nok ? bias[n] : 0.0f;
            #pragma unroll
            for (int r = 0; r < 8; ++r) {
                const int m = m0 + wr * 64 + mt * 16 + rofs + r;
                float v = acc[mt][nt][r] + bv;
                if (RELU) v = fmaxf(v, 0.0f);
                if (OUTBF16) {
                    Cb[(size_t)m * ldc + n] = (__bf16)v;
                } else if (nok) {
                    Cf[(size_t)m * ldc + n] = v;
                }
            }
        }
    }
}

// ---------------------------------------------------------------------------
// In-place row softmax, one wave32 per row (cols = 1000)
// ---------------------------------------------------------------------------
__global__ __launch_bounds__(256)
void softmax_rows(float* __restrict__ data, int rows, int cols) {
    const int row  = blockIdx.x * 8 + (threadIdx.x >> 5);
    const int lane = threadIdx.x & 31;
    if (row >= rows) return;
    float* p = data + (size_t)row * cols;

    float mx = -3.402823466e38f;
    for (int c = lane; c < cols; c += 32) mx = fmaxf(mx, p[c]);
    #pragma unroll
    for (int off = 16; off > 0; off >>= 1) mx = fmaxf(mx, __shfl_xor(mx, off, 32));

    float sum = 0.0f;
    for (int c = lane; c < cols; c += 32) sum += __expf(p[c] - mx);
    #pragma unroll
    for (int off = 16; off > 0; off >>= 1) sum += __shfl_xor(sum, off, 32);

    const float inv = 1.0f / sum;
    for (int c = lane; c < cols; c += 32) p[c] = __expf(p[c] - mx) * inv;
}

// ---------------------------------------------------------------------------
extern "C" void kernel_launch(void* const* d_in, const int* in_sizes, int n_in,
                              void* d_out, int out_size, void* d_ws, size_t ws_size,
                              hipStream_t stream) {
    (void)in_sizes; (void)n_in; (void)out_size; (void)ws_size;

    constexpr int M = 8192, Kdim = 4096, H = 4096, C = 1000, CP = 1024;

    const float* x  = (const float*)d_in[0];
    const float* w1 = (const float*)d_in[1];
    const float* b1 = (const float*)d_in[2];
    const float* w2 = (const float*)d_in[3];
    const float* b2 = (const float*)d_in[4];
    float* out = (float*)d_out;

    // workspace layout (bytes): xb | w1b | w2b(padded) | hb
    char* ws = (char*)d_ws;
    __bf16* xb  = (__bf16*)(ws);
    __bf16* w1b = (__bf16*)(ws + (size_t)M * Kdim * 2);
    __bf16* w2b = (__bf16*)(ws + (size_t)M * Kdim * 2 + (size_t)H * Kdim * 2);
    __bf16* hb  = (__bf16*)(ws + (size_t)M * Kdim * 2 + (size_t)H * Kdim * 2
                               + (size_t)CP * Kdim * 2);

    // 1) f32 -> bf16 staging (memory-bound, ~16us total)
    {
        size_t n8 = (size_t)M * Kdim / 8;
        cvt8_f32_bf16<<<(unsigned)((n8 + 255) / 256), 256, 0, stream>>>(x, xb, n8);
        n8 = (size_t)H * Kdim / 8;
        cvt8_f32_bf16<<<(unsigned)((n8 + 255) / 256), 256, 0, stream>>>(w1, w1b, n8);
        size_t nt = (size_t)CP * (Kdim / 8);
        cvt8_f32_bf16_padrows<<<(unsigned)((nt + 255) / 256), 256, 0, stream>>>(
            w2, w2b, Kdim / 8, C, CP);
    }

    // 2) hidden = relu(x @ W1^T + b1)  -> bf16
    {
        dim3 grid(H / BN, M / BM);   // 32 x 64
        wmma_gemm_bf16<true, true><<<grid, 128, 0, stream>>>(
            xb, w1b, b1, nullptr, hb, Kdim, H, H);
    }

    // 3) logits = hidden @ W2^T + b2  -> f32 straight into d_out (n < 1000)
    {
        dim3 grid(CP / BN, M / BM);  // 8 x 64
        wmma_gemm_bf16<false, false><<<grid, 128, 0, stream>>>(
            hb, w2b, b2, out, nullptr, H, C, C);
    }

    // 4) softmax in place
    softmax_rows<<<M / 8, 256, 0, stream>>>(out, M, C);
}
